// VectorPoolLocalInterpolateModule_10213432230651
// MI455X (gfx1250) — matrix operand order repaired and checked by reading, not compiled, not run on hardware
//
#include <hip/hip_runtime.h>

// ---------------------------------------------------------------------------
// Problem constants (from the reference setup)
// ---------------------------------------------------------------------------
#define BB   2
#define NN   2048          // support points per batch
#define MM   1024          // query centers per batch
#define GG   27            // grid points per center
#define CC   32            // support feature channels
#define HH   64            // hidden width
#define MGQ  (MM * GG)     // 27648 queries per batch
#define RR   (BB * MGQ)    // 55296 total rows
#define KPAD 64            // Cin = 41 padded to 64 for WMMA K-tiling
#define RADIUS_F  4.8f
#define BN_EPS_F  1e-5f

typedef __attribute__((ext_vector_type(16))) _Float16 v16h;
typedef __attribute__((ext_vector_type(8)))  _Float16 v8h;
typedef __attribute__((ext_vector_type(8)))  float    v8f;

// ---------------------------------------------------------------------------
// Kernel 0: weight padding/conversion to f16 + zero BN stat accumulators
//   W1h[n*64+k] = (k < 41) ? w1[n*41+k] : 0      (B-matrix stored [N][K])
//   W2h[n*64+k] = w2[n*64+k]
// ---------------------------------------------------------------------------
__global__ __launch_bounds__(256)
void prep_kernel(const float* __restrict__ w1, const float* __restrict__ w2,
                 _Float16* __restrict__ W1h, _Float16* __restrict__ W2h,
                 float* __restrict__ stats) {
  const int t = blockIdx.x * 256 + threadIdx.x;
  if (t < 256) stats[t] = 0.0f;             // sum1 | sq1 | sum2 | sq2
  if (t < HH * KPAD) {
    const int n = t / KPAD;
    const int k = t % KPAD;
    W1h[t] = (k < CC + 9) ? (_Float16)w1[n * (CC + 9) + k] : (_Float16)0.0f;
    W2h[t] = (_Float16)w2[t];
  }
}

// ---------------------------------------------------------------------------
// Kernel 1: brute-force 3-NN + IDW interpolation + local coords + pad -> f16
// One thread per query. Support cloud for the block's batch staged in LDS.
// ---------------------------------------------------------------------------
__global__ __launch_bounds__(256)
void knn_build_kernel(const float* __restrict__ sxyz,
                      const float* __restrict__ sfeat,
                      const float* __restrict__ centers,
                      _Float16* __restrict__ Xh) {
  __shared__ float4 spts[NN];                       // 32 KB of 320 KB LDS
  const int tid = threadIdx.x;
  const int blocksPerBatch = MGQ / 256;             // 108
  const int b  = blockIdx.x / blocksPerBatch;
  const int qi = (blockIdx.x % blocksPerBatch) * 256 + tid;

  for (int i = tid; i < NN; i += 256) {
    const float* p = sxyz + (size_t)(b * NN + i) * 3;
    spts[i] = make_float4(p[0], p[1], p[2], 0.0f);
  }
  __syncthreads();

  const int r = b * MGQ + qi;
  const float qx = centers[(size_t)r * 3 + 0];
  const float qy = centers[(size_t)r * 3 + 1];
  const float qz = centers[(size_t)r * 3 + 2];

  float b0 = 3.4e38f, b1v = 3.4e38f, b2v = 3.4e38f;
  int   i0 = 0, i1 = 0, i2 = 0;
  for (int n = 0; n < NN; ++n) {
    const float4 p = spts[n];
    const float dx = qx - p.x, dy = qy - p.y, dz = qz - p.z;
    const float d = dx * dx + dy * dy + dz * dz;
    if (d < b2v) {
      if (d < b0)       { b2v = b1v; i2 = i1; b1v = b0; i1 = i0; b0 = d; i0 = n; }
      else if (d < b1v) { b2v = b1v; i2 = i1; b1v = d;  i1 = n; }
      else              { b2v = d;   i2 = n; }
    }
  }

  float d0 = sqrtf(b0), d1 = sqrtf(b1v), d2 = sqrtf(b2v);
  const bool v0 = (d0 <= RADIUS_F), v1 = (d1 <= RADIUS_F), v2 = (d2 <= RADIUS_F);
  const bool empty = !v0;
  if (!v0) { d0 = 1e8f; i0 = 0; }
  if (!v1) { d1 = 1e8f; i1 = 0; }
  if (!v2) { d2 = 1e8f; i2 = 0; }

  const float r0 = 1.0f / (d0 + 1e-8f);
  const float r1 = 1.0f / (d1 + 1e-8f);
  const float r2 = 1.0f / (d2 + 1e-8f);
  const float inv = 1.0f / fmaxf(r0 + r1 + r2, 1e-8f);
  const float iw0 = r0 * inv, iw1 = r1 * inv, iw2 = r2 * inv;

  const float* f0 = sfeat + (size_t)(b * NN + i0) * CC;
  const float* f1 = sfeat + (size_t)(b * NN + i1) * CC;
  const float* f2 = sfeat + (size_t)(b * NN + i2) * CC;

  _Float16 row[KPAD];
  #pragma unroll
  for (int c = 0; c < CC; ++c)
    row[c] = (_Float16)(iw0 * f0[c] + iw1 * f1[c] + iw2 * f2[c]);

  const float4 p0 = spts[i0], p1 = spts[i1], p2 = spts[i2];
  row[CC + 0] = (_Float16)(qx - p0.x); row[CC + 1] = (_Float16)(qy - p0.y); row[CC + 2] = (_Float16)(qz - p0.z);
  row[CC + 3] = (_Float16)(qx - p1.x); row[CC + 4] = (_Float16)(qy - p1.y); row[CC + 5] = (_Float16)(qz - p1.z);
  row[CC + 6] = (_Float16)(qx - p2.x); row[CC + 7] = (_Float16)(qy - p2.y); row[CC + 8] = (_Float16)(qz - p2.z);
  #pragma unroll
  for (int k = CC + 9; k < KPAD; ++k) row[k] = (_Float16)0.0f;
  if (empty) {
    #pragma unroll
    for (int k = 0; k < CC + 9; ++k) row[k] = (_Float16)0.0f;
  }

  v8h* dst = (v8h*)(Xh + (size_t)r * KPAD);
  const v8h* src = (const v8h*)row;
  #pragma unroll
  for (int t = 0; t < 8; ++t) dst[t] = src[t];     // 128 B contiguous per row
}

// ---------------------------------------------------------------------------
// Kernel 2/4: Y[R,64] = X[R,64](f16) * W[64,64]^T via v_wmma_f32_16x16x32_f16
// + per-channel sum / sum-of-squares for training-mode BN stats.
// Block = 256 threads = 8 waves; wave owns 16 rows x 64 cols (4 WMMA accums).
// ---------------------------------------------------------------------------
__global__ __launch_bounds__(256)
void gemm_bnstat_kernel(const _Float16* __restrict__ X,
                        const _Float16* __restrict__ W,
                        float* __restrict__ Y,
                        float* __restrict__ gsum, float* __restrict__ gsq) {
  __shared__ float ssum[HH];
  __shared__ float ssq[HH];
  const int tid = threadIdx.x;
  if (tid < HH) { ssum[tid] = 0.0f; ssq[tid] = 0.0f; }
  __syncthreads();

  const int wave = tid >> 5;
  const int lane = tid & 31;
  const int lrow = lane & 15;         // row-in-tile (A) / col-in-tile (B)
  const int hi   = lane >> 4;         // half-wave select per ISA lane layout
  const int rowbase = blockIdx.x * 128 + wave * 16;
  const int r = rowbase + lrow;

  v8f acc[4] = {};                    // 4 N-tiles of 16 cols

  #pragma unroll
  for (int kt = 0; kt < 2; ++kt) {
    // A fragment, 16-bit 16x32 layout: lanes 0-15 hold K = [0..7]&[16..23],
    // lanes 16-31 hold K = [8..15]&[24..31] -> two contiguous b128 loads.
    const int ka0 = kt * 32 + hi * 8;
    const int ka1 = kt * 32 + 16 + hi * 8;
    const v8h alo = *(const v8h*)(X + (size_t)r * KPAD + ka0);
    const v8h ahi = *(const v8h*)(X + (size_t)r * KPAD + ka1);
    v16h a;
    #pragma unroll
    for (int i = 0; i < 8; ++i) { a[i] = alo[i]; a[8 + i] = ahi[i]; }

    // B fragment, 32x16: lanes 0-15 hold K=[0..7]&[16..23], lanes 16-31 K=[8..15]&[24..31].
    const int kb = kt * 32 + hi * 8;
    #pragma unroll
    for (int nt = 0; nt < 4; ++nt) {
      const int col = nt * 16 + lrow;
      const v8h blo = *(const v8h*)(W + (size_t)col * KPAD + kb);
      const v8h bhi = *(const v8h*)(W + (size_t)col * KPAD + kb + 16);
      v16h bm;
      #pragma unroll
      for (int i = 0; i < 8; ++i) { bm[i] = blo[i]; bm[8 + i] = bhi[i]; }
      acc[nt] = __builtin_amdgcn_wmma_f32_16x16x32_f16(
          /*neg_a=*/false, a, /*neg_b=*/false, bm,
          /*c_mod=*/(short)0, acc[nt], /*reuse_a=*/false, /*reuse_b=*/false);
    }
  }

  // Write D tiles (f32 C/D layout: VGPR g -> row g + 8*hi, lane -> col) and
  // fold per-channel partial stats through LDS before hitting global atomics.
  #pragma unroll
  for (int nt = 0; nt < 4; ++nt) {
    const int col = nt * 16 + lrow;
    float s = 0.0f, s2 = 0.0f;
    #pragma unroll
    for (int g = 0; g < 8; ++g) {
      const float v = acc[nt][g];
      const int m = g + hi * 8;
      Y[(size_t)(rowbase + m) * HH + col] = v;
      s += v; s2 += v * v;
    }
    atomicAdd(&ssum[col], s);
    atomicAdd(&ssq[col], s2);
  }
  __syncthreads();
  if (tid < HH) {
    atomicAdd(&gsum[tid], ssum[tid]);
    atomicAdd(&gsq[tid],  ssq[tid]);
  }
}

// ---------------------------------------------------------------------------
// Kernel 3: BN (training stats) + ReLU, f32 -> f16 activations
// ---------------------------------------------------------------------------
__global__ __launch_bounds__(256)
void bn_relu_h_kernel(const float* __restrict__ Y,
                      const float* __restrict__ sum, const float* __restrict__ sq,
                      const float* __restrict__ gamma, const float* __restrict__ beta,
                      _Float16* __restrict__ Out) {
  const int e = blockIdx.x * 256 + threadIdx.x;
  const int c = e & (HH - 1);
  const float invR = 1.0f / (float)RR;
  const float mu  = sum[c] * invR;
  const float var = sq[c] * invR - mu * mu;
  const float sc  = gamma[c] * rsqrtf(var + BN_EPS_F);
  const float v   = (Y[e] - mu) * sc + beta[c];
  Out[e] = (_Float16)fmaxf(v, 0.0f);
}

// ---------------------------------------------------------------------------
// Kernel 5: BN + ReLU in place on the f32 output
// ---------------------------------------------------------------------------
__global__ __launch_bounds__(256)
void bn_relu_f_kernel(float* __restrict__ Y,
                      const float* __restrict__ sum, const float* __restrict__ sq,
                      const float* __restrict__ gamma, const float* __restrict__ beta) {
  const int e = blockIdx.x * 256 + threadIdx.x;
  const int c = e & (HH - 1);
  const float invR = 1.0f / (float)RR;
  const float mu  = sum[c] * invR;
  const float var = sq[c] * invR - mu * mu;
  const float sc  = gamma[c] * rsqrtf(var + BN_EPS_F);
  const float v   = (Y[e] - mu) * sc + beta[c];
  Y[e] = fmaxf(v, 0.0f);
}

// ---------------------------------------------------------------------------
// Host-side launch
// ---------------------------------------------------------------------------
extern "C" void kernel_launch(void* const* d_in, const int* in_sizes, int n_in,
                              void* d_out, int out_size, void* d_ws, size_t ws_size,
                              hipStream_t stream) {
  (void)in_sizes; (void)n_in; (void)out_size; (void)ws_size;
  const float* sxyz    = (const float*)d_in[0];   // support_xyz  [B*N,3]
  const float* sfeat   = (const float*)d_in[1];   // support_features [B*N,C]
  const float* centers = (const float*)d_in[4];   // new_xyz_grid_centers [B*M,G,3]
  const float* w1 = (const float*)d_in[6];
  const float* g1 = (const float*)d_in[7];
  const float* b1 = (const float*)d_in[8];
  const float* w2 = (const float*)d_in[9];
  const float* g2 = (const float*)d_in[10];
  const float* b2 = (const float*)d_in[11];
  float* out = (float*)d_out;

  char* ws = (char*)d_ws;
  size_t off = 0;
  auto alloc = [&](size_t bytes) -> void* {
    void* p = ws + off;
    off = (off + bytes + 255) & ~(size_t)255;
    return p;
  };
  _Float16* Xh  = (_Float16*)alloc((size_t)RR * KPAD * sizeof(_Float16)); // ~7.1 MB
  _Float16* Y1h = (_Float16*)alloc((size_t)RR * HH  * sizeof(_Float16));  // ~7.1 MB
  float*    Y1f = (float*)   alloc((size_t)RR * HH  * sizeof(float));     // ~14.2 MB
  _Float16* W1h = (_Float16*)alloc((size_t)HH * KPAD * sizeof(_Float16));
  _Float16* W2h = (_Float16*)alloc((size_t)HH * KPAD * sizeof(_Float16));
  float*    stats = (float*) alloc(256 * sizeof(float));
  float *sum1 = stats, *sq1 = stats + 64, *sum2 = stats + 128, *sq2 = stats + 192;

  prep_kernel<<<16, 256, 0, stream>>>(w1, w2, W1h, W2h, stats);
  knn_build_kernel<<<RR / 256, 256, 0, stream>>>(sxyz, sfeat, centers, Xh);
  gemm_bnstat_kernel<<<RR / 128, 256, 0, stream>>>(Xh, W1h, Y1f, sum1, sq1);
  bn_relu_h_kernel<<<(RR * HH) / 256, 256, 0, stream>>>(Y1f, sum1, sq1, g1, b1, Y1h);
  gemm_bnstat_kernel<<<RR / 128, 256, 0, stream>>>(Y1h, W2h, out, sum2, sq2);
  bn_relu_f_kernel<<<(RR * HH) / 256, 256, 0, stream>>>(out, sum2, sq2, g2, b2);
}
